// DQSN_48747878809630
// MI455X (gfx1250) — matrix operand order repaired and checked by reading, not compiled, not run on hardware
//
#include <hip/hip_runtime.h>

// ---------------------------------------------------------------------------
// Fully-fused spiking-MLP for MI455X (gfx1250, wave32, WMMA 16x16x32 f16).
//
// Linearity collapse:  v_lif(T) = S @ w2^T + b2*(1 - 2^-16),
//   S = sum_t 2^(t-17) * spike_t,   spikes from the IF recurrence on
//   h_in = x @ w1^T + b1  (time-invariant drive).  16 GEMMs -> 1 GEMM.
//
// One fused kernel per 32-row batch tile (256 blocks, ~1 per WGP):
//   phase 0: stage x tile (f32->f16) into LDS          (x read once from HBM)
//   phase 1: GEMM1 via WMMA, 2 row-tiles share each B fragment
//            + 16-step IF recurrence in registers -> S tile into LDS (64 KB)
//   phase 2: GEMM2 via WMMA (A from LDS S tile, B from L2-resident w2h)
// HBM traffic ~ x(8MB) + out(8MB) + weights(3MB); L2 weight re-reads 256MB.
// ---------------------------------------------------------------------------

typedef __attribute__((ext_vector_type(16))) _Float16 v16h;
typedef __attribute__((ext_vector_type(8)))  _Float16 v8h;
typedef __attribute__((ext_vector_type(4)))  _Float16 v4h;
typedef __attribute__((ext_vector_type(8)))  float    v8f;

#define IDIM   256
#define HDIM   1024
#define ODIM   256
#define TSTEPS 16
#define MROWS  32   // batch rows per block (2 row-tiles of 16)

// ---- WMMA fragment loaders (CDNA5 16-bit layouts, ISA 7.12.2) --------------
// A (16x32, MxK) from row-major LDS, leading dim ld:
//   lane = m + 16*h ; slots 0..7 = K[8h..8h+7], slots 8..15 = K[16+8h..]
__device__ __forceinline__ v16h lds_a_frag(const _Float16* __restrict__ base,
                                           int ld) {
  const int lane = (int)(threadIdx.x & 31u);
  const int m = lane & 15;
  const int h = lane >> 4;
  const _Float16* p = base + m * ld + 8 * h;
  v8h lo = *(const v8h*)(p);        // ds_load_b128
  v8h hi = *(const v8h*)(p + 16);   // ds_load_b128
  return __builtin_shufflevector(lo, hi, 0, 1, 2, 3, 4, 5, 6, 7,
                                 8, 9, 10, 11, 12, 13, 14, 15);
}

// B (32x16, KxN), B[k][n] = W[n][k], W row-major f16 in global (L2-resident):
//   lane = n + 16*h ; slots 0..7 = K[8h..8h+7], slots 8..15 = K[16+8h..].
__device__ __forceinline__ v16h glb_b_frag(const _Float16* __restrict__ w,
                                           int ld, int n0, int k0) {
  const int lane = (int)(threadIdx.x & 31u);
  const int n = lane & 15;
  const int h = lane >> 4;
  const _Float16* p = w + (size_t)(n0 + n) * ld + k0 + 8 * h;
  v8h lo = *(const v8h*)(p);
  v8h hi = *(const v8h*)(p + 16);
  return __builtin_shufflevector(lo, hi, 0, 1, 2, 3, 4, 5, 6, 7,
                                 8, 9, 10, 11, 12, 13, 14, 15);
}

// ---- f32 -> f16 weight conversion (once per call, L2-resident after) -------
__global__ void cvt_f32_to_f16(const float* __restrict__ src,
                               _Float16* __restrict__ dst, int n) {
  int i = (int)(blockIdx.x * blockDim.x + threadIdx.x) * 4;
  if (i + 3 < n) {
    float4 v = *(const float4*)(src + i);
    dst[i + 0] = (_Float16)v.x;
    dst[i + 1] = (_Float16)v.y;
    dst[i + 2] = (_Float16)v.z;
    dst[i + 3] = (_Float16)v.w;
  }
}

// ---- Fused kernel ----------------------------------------------------------
__global__ __launch_bounds__(256) void snn_fused_kernel(
    const float* __restrict__ x,       // [B, 256] f32
    const _Float16* __restrict__ w1h,  // [1024, 256] f16 row-major
    const float* __restrict__ b1,      // [1024]
    const _Float16* __restrict__ w2h,  // [256, 1024] f16 row-major
    const float* __restrict__ b2,      // [256]
    float* __restrict__ out) {         // [B, 256] f32
  __shared__ _Float16 xs[MROWS * IDIM];  // 16 KB: x tile in f16
  __shared__ _Float16 Sl[MROWS * HDIM];  // 64 KB: weighted-spike tile

  const int tid  = (int)threadIdx.x;
  const int wave = tid >> 5;
  const int lane = tid & 31;
  const int row0 = (int)blockIdx.x * MROWS;
  const int cn   = lane & 15;          // N within a C/D tile
  const int mh   = (lane >> 4) << 3;   // row offset 0 or 8 (C/D layout)

  // ---- phase 0: stage x tile into LDS as f16 (coalesced, read-once) ----
  {
    const float* xsrc = x + (size_t)row0 * IDIM;
#pragma unroll
    for (int i = 0; i < (MROWS * IDIM) / (256 * 4); ++i) {  // 8 iters
      const int idx = (tid + i * 256) * 4;
      float4 v = *(const float4*)(xsrc + idx);
      v4h hv;
      hv[0] = (_Float16)v.x; hv[1] = (_Float16)v.y;
      hv[2] = (_Float16)v.z; hv[3] = (_Float16)v.w;
      *(v4h*)(&xs[idx]) = hv;          // ds_store_b64
    }
  }
  __syncthreads();

  // ---- phase 1: h_in = x @ w1^T + b1 ; wave owns 128 H-cols x 32 rows ----
  const int n0 = wave << 7;
  v8f acc1[2][8] = {};
#pragma unroll
  for (int k = 0; k < IDIM; k += 32) {
    v16h a0 = lds_a_frag(xs + k, IDIM);
    v16h a1 = lds_a_frag(xs + 16 * IDIM + k, IDIM);
#pragma unroll
    for (int n = 0; n < 8; ++n) {
      v16h b = glb_b_frag(w1h, IDIM, n0 + (n << 4), k);  // shared by 2 wmma
      acc1[0][n] = __builtin_amdgcn_wmma_f32_16x16x32_f16(
          false, a0, false, b, (short)0, acc1[0][n], false, false);
      acc1[1][n] = __builtin_amdgcn_wmma_f32_16x16x32_f16(
          false, a1, false, b, (short)0, acc1[1][n], false, false);
    }
  }

  // IF recurrence (integrate -> fire -> hard reset), weighted spike sum,
  // S tile -> LDS for phase 2.
#pragma unroll
  for (int mt = 0; mt < 2; ++mt) {
#pragma unroll
    for (int n = 0; n < 8; ++n) {
      const int col = n0 + (n << 4) + cn;
      const float bias = b1[col];
#pragma unroll
      for (int r = 0; r < 8; ++r) {
        const float hv = acc1[mt][n][r] + bias;
        float v = 0.f, s = 0.f;
        float c = 1.f / 65536.f;  // 2^-16 weight for t=1, doubles each step
#pragma unroll
        for (int t = 0; t < TSTEPS; ++t) {
          v += hv;
          const float sp = (v >= 1.f) ? 1.f : 0.f;  // heaviside(v - V_TH)
          v -= v * sp;                              // hard reset
          s = fmaf(sp, c, s);
          c += c;
        }
        Sl[(mt * 16 + mh + r) * HDIM + col] = (_Float16)s;
      }
    }
  }
  __syncthreads();

  // ---- phase 2: v_lif = S @ w2^T + b2*(1-2^-16) ; wave owns 32 O-cols ----
  const int o0 = wave << 5;
  v8f acc2[2][2] = {};
#pragma unroll
  for (int k = 0; k < HDIM; k += 32) {
    v16h a0 = lds_a_frag(Sl + k, HDIM);
    v16h a1 = lds_a_frag(Sl + 16 * HDIM + k, HDIM);
#pragma unroll
    for (int n = 0; n < 2; ++n) {
      v16h b = glb_b_frag(w2h, HDIM, o0 + (n << 4), k);  // shared by 2 wmma
      acc2[0][n] = __builtin_amdgcn_wmma_f32_16x16x32_f16(
          false, a0, false, b, (short)0, acc2[0][n], false, false);
      acc2[1][n] = __builtin_amdgcn_wmma_f32_16x16x32_f16(
          false, a1, false, b, (short)0, acc2[1][n], false, false);
    }
  }

  const float bscale = 1.f - 1.f / 65536.f;  // sum_t 2^(t-17)
#pragma unroll
  for (int mt = 0; mt < 2; ++mt) {
#pragma unroll
    for (int n = 0; n < 2; ++n) {
      const int col = o0 + (n << 4) + cn;
      const float bias = b2[col] * bscale;
#pragma unroll
      for (int r = 0; r < 8; ++r) {
        out[(size_t)(row0 + mt * 16 + mh + r) * ODIM + col] =
            acc2[mt][n][r] + bias;
      }
    }
  }
}

// ---------------------------------------------------------------------------
extern "C" void kernel_launch(void* const* d_in, const int* in_sizes, int n_in,
                              void* d_out, int out_size, void* d_ws,
                              size_t ws_size, hipStream_t stream) {
  const float* x  = (const float*)d_in[0];  // [B, 256]
  const float* w1 = (const float*)d_in[1];  // [1024, 256]
  const float* b1 = (const float*)d_in[2];  // [1024]
  const float* w2 = (const float*)d_in[3];  // [256, 1024]
  const float* b2 = (const float*)d_in[4];  // [256]
  float* out = (float*)d_out;

  const int Bn = in_sizes[0] / IDIM;  // 8192

  // Workspace: f16 weights only (~1 MB).
  char* ws = (char*)d_ws;
  _Float16* w1h = (_Float16*)ws;  ws += (size_t)HDIM * IDIM * sizeof(_Float16);
  _Float16* w2h = (_Float16*)ws;

  const int n1 = HDIM * IDIM;
  const int n2 = ODIM * HDIM;
  cvt_f32_to_f16<<<(n1 / 4 + 255) / 256, 256, 0, stream>>>(w1, w1h, n1);
  cvt_f32_to_f16<<<(n2 / 4 + 255) / 256, 256, 0, stream>>>(w2, w2h, n2);

  snn_fused_kernel<<<Bn / MROWS, 256, 0, stream>>>(x, w1h, b1, w2h, b2, out);
}
